// GNNEncoder_58858231824537
// MI455X (gfx1250) — compile-verified, matrix operand order checked by reading, not run on hardware
//
#include <hip/hip_runtime.h>
#include <math.h>

#define BATCH 4
#define NV 256
#define NH 128
#define TT 64
#define NL 2
#define EPSV 1e-5f

typedef __attribute__((ext_vector_type(16))) _Float16 v16h;
typedef __attribute__((ext_vector_type(8)))  _Float16 v8h;
typedef __attribute__((ext_vector_type(8)))  float    v8f;

// CDNA5 16-bit WMMA A/B fragment loader from a 16x128 f16 tile (row stride 128).
// Lane ln supplies row (ln&15); K elements for chunk kc: [hi*8 + kc*32 .. +7] and
// [16 + hi*8 + kc*32 .. +7]  (hi = ln>>4). Each half of the v16h is K-contiguous,
// so two 16B LDS loads fill the fragment.
__device__ __forceinline__ v16h frag_from_rows(const _Float16* base, int lane, int kc) {
  const _Float16* r = base + (lane & 15) * NH + ((lane >> 4) << 3) + (kc << 5);
  v8h lo = *(const v8h*)(r);
  v8h hi = *(const v8h*)(r + 16);
  v16h f;
#pragma unroll
  for (int t = 0; t < 8; ++t) { f[t] = lo[t]; f[t + 8] = hi[t]; }
  return f;
}

__device__ __forceinline__ v8f wmma16(v16h a, v16h b, v8f c) {
  return __builtin_amdgcn_wmma_f32_16x16x32_f16(false, a, false, b, (short)0, c, false, false);
}

// sum across aligned 8-lane groups (wave32)
__device__ __forceinline__ float red8(float v) {
  v += __shfl_xor(v, 1);
  v += __shfl_xor(v, 2);
  v += __shfl_xor(v, 4);
  return v;
}

// ---------------- node embedding gather ----------------
__global__ void k_gather(const int* __restrict__ x, const float* __restrict__ emb,
                         float* __restrict__ h) {
  int idx = blockIdx.x * 256 + threadIdx.x;  // B*V*H
  int row = idx >> 7, k = idx & 127;
  h[idx] = emb[(size_t)x[row] * NH + k];
}

// ---------------- timestep MLP + per-layer te projection ----------------
__global__ __launch_bounds__(128) void k_time(const int* __restrict__ ts,
    const float* __restrict__ t1w, const float* __restrict__ t1b,
    const float* __restrict__ t2w, const float* __restrict__ t2b,
    const float* __restrict__ tlw, const float* __restrict__ tlb,
    float* __restrict__ teproj /* (L,B,H) */) {
  __shared__ float emb[BATCH][NH];
  __shared__ float r1[BATCH][TT];
  __shared__ float r2[BATCH][TT];
  int t = threadIdx.x;
  if (t < TT) {
    float freq = __expf(-logf(10000.0f) * (float)t / (float)TT);
    for (int b = 0; b < BATCH; ++b) {
      float arg = (float)ts[b] * freq;
      emb[b][t] = __cosf(arg);
      emb[b][TT + t] = __sinf(arg);
    }
  }
  __syncthreads();
  if (t < TT) {
    for (int b = 0; b < BATCH; ++b) {
      float acc = t1b[t];
      for (int d = 0; d < NH; ++d) acc += emb[b][d] * t1w[d * TT + t];
      r1[b][t] = fmaxf(acc, 0.0f);
    }
  }
  __syncthreads();
  if (t < TT) {
    for (int b = 0; b < BATCH; ++b) {
      float acc = t2b[t];
      for (int d = 0; d < TT; ++d) acc += r1[b][d] * t2w[d * TT + t];
      r2[b][t] = fmaxf(acc, 0.0f);  // relu(te)
    }
  }
  __syncthreads();
  for (int l = 0; l < NL; ++l)
    for (int b = 0; b < BATCH; ++b) {
      float acc = tlb[l * NH + t];
      for (int d = 0; d < TT; ++d) acc += r2[b][d] * tlw[(l * TT + d) * NH + t];
      teproj[(l * BATCH + b) * NH + t] = acc;
    }
}

// ---------------- e0 = sincos-embed(graph) @ edge_w + edge_b (WMMA) ----------------
__global__ __launch_bounds__(128) void k_edge_embed(const float* __restrict__ graph,
    const float* __restrict__ edge_w, const float* __restrict__ edge_b,
    float* __restrict__ e) {
  __shared__ _Float16 wT[NH * NH];   // transposed weights: wT[n][k]
  __shared__ float inv_dimt[NH];
  __shared__ float eb[NH];
  int tid = threadIdx.x;
  for (int idx = tid; idx < NH * NH; idx += 128) {
    int k = idx >> 7, n = idx & 127;
    wT[n * NH + k] = (_Float16)edge_w[idx];
  }
  if (tid < NH) {
    inv_dimt[tid] = __expf(-logf(10000.0f) * (2.0f * (float)(tid >> 1) / (float)NH));
    eb[tid] = edge_b[tid];
  }
  __syncthreads();

  int wave = tid >> 5, lane = tid & 31;
  int tile = blockIdx.x * 4 + wave;  // 16-row tile over B*V*V rows
  int row0 = tile * 16;
  float g = graph[row0 + (lane & 15)];

  v16h a[4];
#pragma unroll
  for (int kc = 0; kc < 4; ++kc) {
#pragma unroll
    for (int p = 0; p < 16; ++p) {
      int h = ((p & 8) << 1) + ((lane >> 4) << 3) + (p & 7) + (kc << 5);
      float pos = g * inv_dimt[h];
      a[kc][p] = (_Float16)((h & 1) ? __cosf(pos) : __sinf(pos));
    }
  }
  int hi = lane >> 4, col = lane & 15;
#pragma unroll
  for (int nt = 0; nt < 8; ++nt) {
    v8f c = {};
#pragma unroll
    for (int kc = 0; kc < 4; ++kc)
      c = wmma16(a[kc], frag_from_rows(wT + nt * 16 * NH, lane, kc), c);
    int n = nt * 16 + col;
    float bn = eb[n];
#pragma unroll
    for (int r = 0; r < 8; ++r)
      e[(size_t)(row0 + r + 8 * hi) * NH + n] = c[r] + bn;
  }
}

// ---------------- node linear (one of U/V/A/B) : out = h @ W + b (WMMA) ----------------
__global__ __launch_bounds__(128) void k_node_lin(const float* __restrict__ hbuf,
    const float* __restrict__ W, const float* __restrict__ bias,
    float* __restrict__ out) {
  __shared__ _Float16 wT[NH * NH];
  __shared__ _Float16 hT[4][16 * NH];
  __shared__ float bs[NH];
  int tid = threadIdx.x;
  for (int idx = tid; idx < NH * NH; idx += 128) {
    int k = idx >> 7, n = idx & 127;
    wT[n * NH + k] = (_Float16)W[idx];
  }
  if (tid < NH) bs[tid] = bias[tid];
  int wave = tid >> 5, lane = tid & 31;
  int tile = blockIdx.x * 4 + wave;
  int row0 = tile * 16;
  for (int idx = lane; idx < 16 * NH; idx += 32)
    hT[wave][idx] = (_Float16)hbuf[(size_t)row0 * NH + idx];
  __syncthreads();
  v16h a[4];
#pragma unroll
  for (int kc = 0; kc < 4; ++kc) a[kc] = frag_from_rows(hT[wave], lane, kc);
  int hi = lane >> 4, col = lane & 15;
#pragma unroll
  for (int nt = 0; nt < 8; ++nt) {
    v8f c = {};
#pragma unroll
    for (int kc = 0; kc < 4; ++kc)
      c = wmma16(a[kc], frag_from_rows(wT + nt * 16 * NH, lane, kc), c);
    int n = nt * 16 + col;
#pragma unroll
    for (int r = 0; r < 8; ++r)
      out[(size_t)(row0 + r + 8 * hi) * NH + n] = c[r] + bs[n];
  }
}

// ------------- fused per-layer edge pipeline: one block per (b,i) -------------
__global__ __launch_bounds__(128) void k_layer_edge(
    float* __restrict__ e,
    const float* __restrict__ linA, const float* __restrict__ linB,
    const float* __restrict__ linV,
    const float* __restrict__ Cw, const float* __restrict__ Cb,
    const float* __restrict__ plw, const float* __restrict__ plb,
    const float* __restrict__ ne_g, const float* __restrict__ ne_b,
    const float* __restrict__ pl_g, const float* __restrict__ pl_b,
    const float* __restrict__ teproj, float* __restrict__ agg_out) {
  extern __shared__ char smem[];
  _Float16* CwT = (_Float16*)smem;              // 32768 B
  _Float16* PwT = (_Float16*)(smem + 32768);    // 32768 B
  float* Ein = (float*)(smem + 65536);          // 8192 B (e_in tile, residual)
  float* EN  = (float*)(smem + 73728);          // 8192 B (e_new staging)
  float* Ahs = (float*)(smem + 81920);          // 8192 B
  float* Vhs = (float*)(smem + 90112);          // 8192 B
  _Float16* EH = (_Float16*)(smem + 98304);     // 4096 B (f16 GEMM input)
  float* Bhi = (float*)(smem + 102400);
  float* tep = (float*)(smem + 102912);
  float* cbs = (float*)(smem + 103424);
  float* pbs = (float*)(smem + 103936);
  float* ng  = (float*)(smem + 104448);
  float* nb  = (float*)(smem + 104960);
  float* pg  = (float*)(smem + 105472);
  float* pb  = (float*)(smem + 105984);
  float* agg = (float*)(smem + 106496);         // total 107008 B

  int tid = threadIdx.x;
  int b = blockIdx.x >> 8;
  int i = blockIdx.x & 255;

  for (int idx = tid; idx < NH * NH; idx += 128) {
    int k = idx >> 7, n = idx & 127;
    CwT[n * NH + k] = (_Float16)Cw[idx];
    PwT[n * NH + k] = (_Float16)plw[idx];
  }
  if (tid < NH) {
    Bhi[tid] = linB[(size_t)(b * NV + i) * NH + tid];
    tep[tid] = teproj[b * NH + tid];
    cbs[tid] = Cb[tid]; pbs[tid] = plb[tid];
    ng[tid] = ne_g[tid]; nb[tid] = ne_b[tid];
    pg[tid] = pl_g[tid]; pb[tid] = pl_b[tid];
    agg[tid] = 0.0f;
  }
  __syncthreads();

  int wave = tid >> 5, lane = tid & 31;
  int hi = lane >> 4, col = lane & 15;
  size_t ebase = (size_t)(b * NV + i) * NV * NH;

  for (int jt = 0; jt < 16; ++jt) {
    int j0 = jt * 16;
    for (int idx = tid; idx < 16 * NH; idx += 128) {   // stage tile
      int row = idx >> 7, k = idx & 127;
      float v = e[ebase + (size_t)(j0 + row) * NH + k];
      Ein[idx] = v;
      EH[idx] = (_Float16)v;
      Ahs[idx] = linA[(size_t)(b * NV + j0 + row) * NH + k];
      Vhs[idx] = linV[(size_t)(b * NV + j0 + row) * NH + k];
    }
    __syncthreads();

    // prefetch next j-tile while we compute on this one (global_prefetch_b8)
    if (jt < 15) {
      int jn = j0 + 16;
      __builtin_prefetch(&e[ebase + (size_t)jn * NH + tid * 16], 0, 3);
      __builtin_prefetch(&linA[(size_t)(b * NV + jn) * NH + tid * 16], 0, 3);
      __builtin_prefetch(&linV[(size_t)(b * NV + jn) * NH + tid * 16], 0, 3);
    }

    v16h a[4];
#pragma unroll
    for (int kc = 0; kc < 4; ++kc) a[kc] = frag_from_rows(EH, lane, kc);

    // GEMM1 (Ce) + gate + agg, wave w owns output column-tiles {w, w+4}
#pragma unroll
    for (int hf = 0; hf < 2; ++hf) {
      int nt = wave + hf * 4;
      v8f c = {};
#pragma unroll
      for (int kc = 0; kc < 4; ++kc)
        c = wmma16(a[kc], frag_from_rows(CwT + nt * 16 * NH, lane, kc), c);
      int n = nt * 16 + col;
      float aggp = 0.0f;
#pragma unroll
      for (int r = 0; r < 8; ++r) {
        int row = r + 8 * hi;
        float e1 = c[r] + cbs[n] + Ahs[row * NH + n] + Bhi[n];
        EN[row * NH + n] = e1;
        float gate = 1.0f / (1.0f + __expf(-e1));
        aggp += gate * Vhs[row * NH + n];
      }
      aggp += __shfl_xor(aggp, 16);
      if (hi == 0) agg[n] += aggp;   // single writer per n: no atomics needed
    }
    __syncthreads();

    // per-row LN -> relu -> +te -> LN -> SiLU : 8 threads/row, values in registers
    {
      int row = tid >> 3, sub = tid & 7;
      const float* rp = EN + row * NH;
      float vals[16];
#pragma unroll
      for (int t = 0; t < 16; ++t) vals[t] = rp[sub + 8 * t];
      float s = 0.0f;
#pragma unroll
      for (int t = 0; t < 16; ++t) s += vals[t];
      float m = red8(s) * (1.0f / NH);
      float vv = 0.0f;
#pragma unroll
      for (int t = 0; t < 16; ++t) { float d = vals[t] - m; vv += d * d; }
      float rs = rsqrtf(red8(vv) * (1.0f / NH) + EPSV);
      float s2 = 0.0f;
#pragma unroll
      for (int t = 0; t < 16; ++t) {
        int k = sub + 8 * t;
        float u = fmaxf((vals[t] - m) * rs * ng[k] + nb[k], 0.0f) + tep[k];
        vals[t] = u; s2 += u;
      }
      float m2 = red8(s2) * (1.0f / NH);
      float v2 = 0.0f;
#pragma unroll
      for (int t = 0; t < 16; ++t) { float d = vals[t] - m2; v2 += d * d; }
      float rs2 = rsqrtf(red8(v2) * (1.0f / NH) + EPSV);
      _Float16* oh = EH + row * NH;
#pragma unroll
      for (int t = 0; t < 16; ++t) {
        int k = sub + 8 * t;
        float ln2 = (vals[t] - m2) * rs2 * pg[k] + pb[k];
        oh[k] = (_Float16)(ln2 / (1.0f + __expf(-ln2)));  // SiLU
      }
    }
    __syncthreads();

    // GEMM2 (plw) + residual store
#pragma unroll
    for (int kc = 0; kc < 4; ++kc) a[kc] = frag_from_rows(EH, lane, kc);
#pragma unroll
    for (int hf = 0; hf < 2; ++hf) {
      int nt = wave + hf * 4;
      v8f c = {};
#pragma unroll
      for (int kc = 0; kc < 4; ++kc)
        c = wmma16(a[kc], frag_from_rows(PwT + nt * 16 * NH, lane, kc), c);
      int n = nt * 16 + col;
#pragma unroll
      for (int r = 0; r < 8; ++r) {
        int row = r + 8 * hi;
        e[ebase + (size_t)(j0 + row) * NH + n] = c[r] + pbs[n] + Ein[row * NH + n];
      }
    }
    __syncthreads();
  }
  if (tid < NH) agg_out[(size_t)(b * NV + i) * NH + tid] = agg[tid];
}

// ---------------- h update: h += relu(LN(Uh + agg)) ----------------
__global__ __launch_bounds__(128) void k_node_update(float* __restrict__ h,
    const float* __restrict__ Uh, const float* __restrict__ agg,
    const float* __restrict__ gam, const float* __restrict__ bet) {
  __shared__ float red[128];
  __shared__ float red2[128];
  int row = blockIdx.x, t = threadIdx.x;
  float v = Uh[(size_t)row * NH + t] + agg[(size_t)row * NH + t];
  red[t] = v; red2[t] = v * v;
  __syncthreads();
  for (int s = 64; s > 0; s >>= 1) {
    if (t < s) { red[t] += red[t + s]; red2[t] += red2[t + s]; }
    __syncthreads();
  }
  float m = red[0] * (1.0f / NH);
  float var = red2[0] * (1.0f / NH) - m * m;
  float rs = rsqrtf(var + EPSV);
  h[(size_t)row * NH + t] += fmaxf((v - m) * rs * gam[t] + bet[t], 0.0f);
}

// ---------------- GroupNorm stats over (B,32 groups) ----------------
__global__ __launch_bounds__(256) void k_gn_stats(const float* __restrict__ e,
                                                  float* __restrict__ stats) {
  __shared__ float rsm[256], rss[256];
  int b = blockIdx.x >> 5, g = blockIdx.x & 31, t = threadIdx.x;
  float s = 0.0f, ss = 0.0f;
  for (int ij = t; ij < NV * NV; ij += 256) {
    const float* p = e + ((size_t)b * NV * NV + ij) * NH + g * 4;
    float4 v = *(const float4*)p;
    s += v.x + v.y + v.z + v.w;
    ss += v.x * v.x + v.y * v.y + v.z * v.z + v.w * v.w;
  }
  rsm[t] = s; rss[t] = ss;
  __syncthreads();
  for (int st = 128; st > 0; st >>= 1) {
    if (t < st) { rsm[t] += rsm[t + st]; rss[t] += rss[t + st]; }
    __syncthreads();
  }
  if (t == 0) {
    const float n = (float)(NV * NV * 4);
    float m = rsm[0] / n;
    float var = rss[0] / n - m * m;
    stats[blockIdx.x * 2 + 0] = m;
    stats[blockIdx.x * 2 + 1] = rsqrtf(var + EPSV);
  }
}

// ---------------- fused GN apply + ReLU + 1x1 conv ----------------
__global__ __launch_bounds__(256) void k_out(const float* __restrict__ e,
    const float* __restrict__ stats,
    const float* __restrict__ gg, const float* __restrict__ gb,
    const float* __restrict__ cw, const float* __restrict__ cb,
    float* __restrict__ out) {
  int idx = blockIdx.x * 256 + threadIdx.x;  // b*V*V + ij
  int b = idx >> 16;
  const float* row = e + (size_t)idx * NH;
  float acc = cb[0];
  for (int hh = 0; hh < NH; hh += 4) {
    float4 v = *(const float4*)(row + hh);
    int g = hh >> 2;
    float m = stats[(b * 32 + g) * 2 + 0];
    float r = stats[(b * 32 + g) * 2 + 1];
    float vv[4] = {v.x, v.y, v.z, v.w};
#pragma unroll
    for (int c = 0; c < 4; ++c)
      acc += cw[hh + c] * fmaxf((vv[c] - m) * r * gg[hh + c] + gb[hh + c], 0.0f);
  }
  out[idx] = acc;
}

extern "C" void kernel_launch(void* const* d_in, const int* in_sizes, int n_in,
                              void* d_out, int out_size, void* d_ws, size_t ws_size,
                              hipStream_t stream) {
  (void)in_sizes; (void)n_in; (void)out_size; (void)ws_size;
  const int*   x        = (const int*)d_in[0];
  const float* graph    = (const float*)d_in[1];
  const int*   ts       = (const int*)d_in[2];
  const float* node_emb = (const float*)d_in[3];
  const float* edge_w   = (const float*)d_in[4];
  const float* edge_b   = (const float*)d_in[5];
  const float* t1w = (const float*)d_in[6];  const float* t1b = (const float*)d_in[7];
  const float* t2w = (const float*)d_in[8];  const float* t2b = (const float*)d_in[9];
  const float* Uw = (const float*)d_in[10];  const float* Ub = (const float*)d_in[11];
  const float* Vw = (const float*)d_in[12];  const float* Vb = (const float*)d_in[13];
  const float* Aw = (const float*)d_in[14];  const float* Ab = (const float*)d_in[15];
  const float* Bw = (const float*)d_in[16];  const float* Bb = (const float*)d_in[17];
  const float* Cw = (const float*)d_in[18];  const float* Cb = (const float*)d_in[19];
  const float* nh_g = (const float*)d_in[20]; const float* nh_b = (const float*)d_in[21];
  const float* ne_g = (const float*)d_in[22]; const float* ne_b = (const float*)d_in[23];
  const float* tl_w = (const float*)d_in[24]; const float* tl_b = (const float*)d_in[25];
  const float* pl_g = (const float*)d_in[26]; const float* pl_b = (const float*)d_in[27];
  const float* plw = (const float*)d_in[28];  const float* plb = (const float*)d_in[29];
  const float* gn_g = (const float*)d_in[30]; const float* gn_b = (const float*)d_in[31];
  const float* conv_w = (const float*)d_in[32]; const float* conv_b = (const float*)d_in[33];

  float* ws  = (float*)d_ws;
  float* e   = ws;                                          // B*V*V*H
  float* lin = e + (size_t)BATCH * NV * NV * NH;            // 4 * B*V*H (U,V,A,B)
  float* hb  = lin + 4 * (size_t)BATCH * NV * NH;           // B*V*H
  float* agg = hb + (size_t)BATCH * NV * NH;                // B*V*H
  float* tep = agg + (size_t)BATCH * NV * NH;               // L*B*H
  float* stats = tep + (size_t)NL * BATCH * NH;             // 256

  k_gather<<<BATCH * NV * NH / 256, 256, 0, stream>>>(x, node_emb, hb);
  k_time<<<1, 128, 0, stream>>>(ts, t1w, t1b, t2w, t2b, tl_w, tl_b, tep);
  k_edge_embed<<<BATCH * NV * NV / 64, 128, 0, stream>>>(graph, edge_w, edge_b, e);

  const float* Wsets[4] = {Uw, Vw, Aw, Bw};
  const float* Bsets[4] = {Ub, Vb, Ab, Bb};
  for (int l = 0; l < NL; ++l) {
    for (int s = 0; s < 4; ++s)
      k_node_lin<<<BATCH * NV / 64, 128, 0, stream>>>(
          hb, Wsets[s] + (size_t)l * NH * NH, Bsets[s] + l * NH,
          lin + (size_t)s * BATCH * NV * NH);
    k_layer_edge<<<BATCH * NV, 128, 107008, stream>>>(
        e,
        lin + 2 * (size_t)BATCH * NV * NH,  // Ah
        lin + 3 * (size_t)BATCH * NV * NH,  // Bh
        lin + 1 * (size_t)BATCH * NV * NH,  // Vh
        Cw + (size_t)l * NH * NH, Cb + l * NH,
        plw + (size_t)l * NH * NH, plb + l * NH,
        ne_g + l * NH, ne_b + l * NH, pl_g + l * NH, pl_b + l * NH,
        tep + (size_t)l * BATCH * NH, agg);
    k_node_update<<<BATCH * NV, 128, 0, stream>>>(hb, lin /*Uh*/, agg,
                                                  nh_g + l * NH, nh_b + l * NH);
  }
  k_gn_stats<<<BATCH * 32, 256, 0, stream>>>(e, stats);
  k_out<<<BATCH * NV * NV / 256, 256, 0, stream>>>(e, stats, gn_g, gn_b,
                                                   conv_w, conv_b, (float*)d_out);
}
